// FullSelfAttention_78855599554924
// MI455X (gfx1250) — matrix operand order repaired and checked by reading, not compiled, not run on hardware
//
#include <hip/hip_runtime.h>
#include <hip/hip_bf16.h>
#include <stdint.h>

// Constants for this problem
// B=4, N=1024, C=1024, NH=16, HD=64

typedef _Float16 v16h __attribute__((ext_vector_type(16)));
typedef _Float16 v8h  __attribute__((ext_vector_type(8)));
typedef float    v8f  __attribute__((ext_vector_type(8)));

__device__ __forceinline__ v8f wmma_f16(v16h a, v16h b, v8f c) {
    // D = A(16x32 f16) * B(32x16 f16) + C(16x16 f32)
    return __builtin_amdgcn_wmma_f32_16x16x32_f16(
        /*neg_a=*/false, a, /*neg_b=*/false, b,
        /*c_mod=*/(short)0, c, /*reuse_a=*/false, /*reuse_b=*/false);
}

// A-fragment (16x32, row-major tile in LDS, base already at [m0][k0], stride ldm halfs).
// ISA layout: lanes 0-15 hold row M=lane, elems 0..7 = K 0..7, elems 8..15 = K 16..23;
// lanes 16-31 hold K 8..15 and 24..31.
__device__ __forceinline__ v16h load_a_frag(const _Float16* base, int ldm, int lane) {
    int r  = lane & 15;
    int kh = (lane & 16) ? 8 : 0;
    const _Float16* p = base + r * ldm + kh;
    union { v16h v; v8h h[2]; } u;
    u.h[0] = *(const v8h*)(p);
    u.h[1] = *(const v8h*)(p + 16);
    return u.v;
}

// B-fragment (32x16). LDS tile stored "n-major": BT[n][k], stride ldk halfs, base at [n0][k0].
// ISA layout: lane holds column n=lane%16, lanes 0-15: K 0..7 and 16..23, lanes 16-31: K 8..15 and 24..31.
__device__ __forceinline__ v16h load_b_frag(const _Float16* base, int ldk, int lane) {
    int n  = lane & 15;
    int kb = (lane & 16) ? 8 : 0;
    const _Float16* p = base + n * ldk + kb;
    union { v16h v; v8h h[2]; } u;
    u.h[0] = *(const v8h*)(p);
    u.h[1] = *(const v8h*)(p + 16);
    return u.v;
}

// ---------------------------------------------------------------------------
// Kernel A: QKV GEMM.  X[4096,1024] fp32 * W_qkv[1024,3072] fp32 + b_qkv
// -> f16 scatter into qkv[3][4][16][1024][64].
// Block: 128 threads (4 waves), tile 64(M) x 64(N), K-chunks of 32.
// ---------------------------------------------------------------------------
__global__ __launch_bounds__(128) void qkv_gemm_kernel(
    const float* __restrict__ X, const float* __restrict__ Wq,
    const float* __restrict__ bq, _Float16* __restrict__ qkv)
{
    __shared__ __align__(16) _Float16 As[64 * 32];   // row-major [m][k]
    __shared__ __align__(16) _Float16 Bt[64 * 32];   // transposed [n][k]
    int t = threadIdx.x, lane = t & 31, w = t >> 5;
    int tileN = blockIdx.x % 48;   // 3072/64
    int tileM = blockIdx.x / 48;   // 4096/64
    int mBase = tileM * 64, nBase = tileN * 64;

    v8f acc[4] = {{}, {}, {}, {}};
    for (int kb = 0; kb < 1024; kb += 32) {
        // A: 64x32 fp32 -> f16 LDS (float4 per thread x4)
        #pragma unroll
        for (int i = 0; i < 4; ++i) {
            int id = t + i * 128;            // 0..511 float4s
            int row = id >> 3, c4 = id & 7;  // 8 float4 per row
            float4 f = *(const float4*)(X + (mBase + row) * 1024 + kb + c4 * 4);
            _Float16* dst = As + row * 32 + c4 * 4;
            dst[0] = (_Float16)f.x; dst[1] = (_Float16)f.y;
            dst[2] = (_Float16)f.z; dst[3] = (_Float16)f.w;
        }
        // B: 32x64 fp32 -> transposed f16 LDS
        #pragma unroll
        for (int i = 0; i < 4; ++i) {
            int id = t + i * 128;
            int kr = id >> 4, c4 = id & 15;  // 16 float4 per row of 64
            float4 f = *(const float4*)(Wq + (kb + kr) * 3072 + nBase + c4 * 4);
            Bt[(c4 * 4 + 0) * 32 + kr] = (_Float16)f.x;
            Bt[(c4 * 4 + 1) * 32 + kr] = (_Float16)f.y;
            Bt[(c4 * 4 + 2) * 32 + kr] = (_Float16)f.z;
            Bt[(c4 * 4 + 3) * 32 + kr] = (_Float16)f.w;
        }
        __syncthreads();
        v16h a = load_a_frag(As + (w * 16) * 32, 32, lane);
        #pragma unroll
        for (int ct = 0; ct < 4; ++ct) {
            v16h b = load_b_frag(Bt + (ct * 16) * 32, 32, lane);
            acc[ct] = wmma_f16(a, b, acc[ct]);
        }
        __syncthreads();
    }
    // Epilogue: bias + scatter to [which][b][h][n][d] f16
    int roff = (lane & 16) ? 8 : 0;
    int nlan = lane & 15;
    #pragma unroll
    for (int ct = 0; ct < 4; ++ct) {
        int col = nBase + ct * 16 + nlan;
        float bias = bq[col];
        int which = col >> 10;
        int rem = col & 1023;
        int h = rem >> 6, d = rem & 63;
        #pragma unroll
        for (int r = 0; r < 8; ++r) {
            int rowg = mBase + w * 16 + roff + r;
            int bb = rowg >> 10, nrow = rowg & 1023;
            float v = acc[ct][r] + bias;
            qkv[(((which * 4 + bb) * 16 + h) * 1024 + nrow) * 64 + d] = (_Float16)v;
        }
    }
}

// ---------------------------------------------------------------------------
// Kernel B: in-place RoPE on q and k halves of the qkv buffer.
// gid -> (sel in {q,k}, b, h, n, d<32); pairs (d, d+32).
// ---------------------------------------------------------------------------
__global__ __launch_bounds__(256) void rope_kernel(
    _Float16* __restrict__ qkv, const float* __restrict__ cosp,
    const float* __restrict__ sinp)
{
    int gid = blockIdx.x * 256 + threadIdx.x;   // < 2*4*16*1024*32 = 4194304
    int d   = gid & 31;
    int n   = (gid >> 5) & 1023;
    int h   = (gid >> 15) & 15;
    int b   = (gid >> 19) & 3;
    int sel = gid >> 21;                        // 0 = q, 1 = k
    _Float16* base = qkv + (((sel * 4 + b) * 16 + h) * 1024 + n) * 64;
    float t1 = (float)base[d], t2 = (float)base[d + 32];
    float c1 = cosp[n * 64 + d],      s1 = sinp[n * 64 + d];
    float c2 = cosp[n * 64 + d + 32], s2 = sinp[n * 64 + d + 32];
    base[d]      = (_Float16)(t1 * c1 - t2 * s1);
    base[d + 32] = (_Float16)(t2 * c2 + t1 * s2);
}

// ---------------------------------------------------------------------------
// Kernel C: flash attention. One block per (b, h, 64-row q block).
// 128 threads / 4 waves; each wave owns a 16-row stripe.
// Online softmax over 16 key blocks of 64.
// ---------------------------------------------------------------------------
__global__ __launch_bounds__(128) void attn_kernel(
    const _Float16* __restrict__ qkv, _Float16* __restrict__ aout)
{
    __shared__ __align__(16) _Float16 Qs[64 * 64];    // [row][d]
    __shared__ __align__(16) _Float16 Ks[64 * 64];    // [key][d]  (== B-frag layout for K^T)
    __shared__ __align__(16) _Float16 VTs[64 * 64];   // [d][key]  (transposed V)
    __shared__ __align__(16) float    Ss[4][16 * 64]; // per-wave S scratch
    __shared__ __align__(16) _Float16 Ps[4][16 * 64]; // per-wave exp(S) f16
    __shared__ float corr[4][16];
    __shared__ float rl[4][16];

    int t = threadIdx.x, lane = t & 31, w = t >> 5;
    int qb = blockIdx.x & 15;
    int h  = (blockIdx.x >> 4) & 15;
    int b  = blockIdx.x >> 8;

    const _Float16* qp = qkv + (size_t)(((0 * 4 + b) * 16 + h)) * (1024 * 64);
    const _Float16* kp = qkv + (size_t)(((1 * 4 + b) * 16 + h)) * (1024 * 64);
    const _Float16* vp = qkv + (size_t)(((2 * 4 + b) * 16 + h)) * (1024 * 64);

    // Load Q block (64x64 f16)
    #pragma unroll
    for (int i = 0; i < 4; ++i) {
        int id = t + i * 128; int row = id >> 3, c8 = id & 7;
        *(uint4*)(Qs + row * 64 + c8 * 8) =
            *(const uint4*)(qp + (qb * 64 + row) * 64 + c8 * 8);
    }
    __syncthreads();

    v8f O[4] = {{}, {}, {}, {}};
    float m_run = -1e30f, l_run = 0.f;
    int roff = (lane & 16) ? 8 : 0;
    int nlan = lane & 15;

    v16h aq0 = load_a_frag(Qs + (w * 16) * 64, 64, lane);        // d = 0..31
    v16h aq1 = load_a_frag(Qs + (w * 16) * 64 + 32, 64, lane);   // d = 32..63

    for (int kb = 0; kb < 1024; kb += 64) {
        // Load K (row-major) and V (transposed) blocks
        #pragma unroll
        for (int i = 0; i < 4; ++i) {
            int id = t + i * 128; int row = id >> 3, c8 = id & 7;
            *(uint4*)(Ks + row * 64 + c8 * 8) =
                *(const uint4*)(kp + (kb + row) * 64 + c8 * 8);
            union { uint4 u; _Float16 hh[8]; } uv;
            uv.u = *(const uint4*)(vp + (kb + row) * 64 + c8 * 8);
            #pragma unroll
            for (int e = 0; e < 8; ++e) VTs[(c8 * 8 + e) * 64 + row] = uv.hh[e];
        }
        __syncthreads();

        // S = (Q K^T) * scale  -> per-wave LDS scratch
        #pragma unroll
        for (int ct = 0; ct < 4; ++ct) {
            v16h b0 = load_b_frag(Ks + (ct * 16) * 64, 64, lane);
            v16h b1 = load_b_frag(Ks + (ct * 16) * 64 + 32, 64, lane);
            v8f s = {};
            s = wmma_f16(aq0, b0, s);
            s = wmma_f16(aq1, b1, s);
            #pragma unroll
            for (int r = 0; r < 8; ++r)
                Ss[w][(roff + r) * 64 + ct * 16 + nlan] = s[r] * 0.125f;
        }
        __syncthreads();

        // Online softmax: lanes 0..15 each own one row of the wave's stripe
        if (lane < 16) {
            float* srow = &Ss[w][lane * 64];
            float mloc = -1e30f;
            #pragma unroll 8
            for (int c = 0; c < 64; ++c) mloc = fmaxf(mloc, srow[c]);
            float mnew = fmaxf(m_run, mloc);
            float cf = __expf(m_run - mnew);
            float ls = l_run * cf;
            _Float16* prow = &Ps[w][lane * 64];
            #pragma unroll 8
            for (int c = 0; c < 64; ++c) {
                float p = __expf(srow[c] - mnew);
                ls += p;
                prow[c] = (_Float16)p;
            }
            m_run = mnew; l_run = ls;
            corr[w][lane] = cf;
        }
        __syncthreads();

        // Rescale O by per-row correction
        float cfr[8];
        #pragma unroll
        for (int r = 0; r < 8; ++r) cfr[r] = corr[w][roff + r];
        #pragma unroll
        for (int ct = 0; ct < 4; ++ct)
            #pragma unroll
            for (int r = 0; r < 8; ++r) O[ct][r] = O[ct][r] * cfr[r];

        // O += P * V
        v16h ap0 = load_a_frag(&Ps[w][0], 64, lane);        // j = 0..31
        v16h ap1 = load_a_frag(&Ps[w][0] + 32, 64, lane);   // j = 32..63
        #pragma unroll
        for (int ct = 0; ct < 4; ++ct) {
            v16h b0 = load_b_frag(VTs + (ct * 16) * 64, 64, lane);
            v16h b1 = load_b_frag(VTs + (ct * 16) * 64 + 32, 64, lane);
            O[ct] = wmma_f16(ap0, b0, O[ct]);
            O[ct] = wmma_f16(ap1, b1, O[ct]);
        }
        __syncthreads();
    }

    // Final normalization and store (layout [B][N][NH][HD] == [4096][1024])
    if (lane < 16) rl[w][lane] = 1.0f / (l_run + 1e-9f);
    __syncthreads();
    float rlr[8];
    #pragma unroll
    for (int r = 0; r < 8; ++r) rlr[r] = rl[w][roff + r];
    #pragma unroll
    for (int ct = 0; ct < 4; ++ct) {
        int d = ct * 16 + nlan;
        #pragma unroll
        for (int r = 0; r < 8; ++r) {
            int n = qb * 64 + w * 16 + roff + r;
            float v = O[ct][r] * rlr[r];
            aout[((b * 1024 + n) * 16 + h) * 64 + d] = (_Float16)v;
        }
    }
}

// ---------------------------------------------------------------------------
// Kernel D: output projection.  Aout[4096,1024] f16 * W_proj[1024,1024] fp32
// + b_proj -> out fp32.
// ---------------------------------------------------------------------------
__global__ __launch_bounds__(128) void proj_gemm_kernel(
    const _Float16* __restrict__ Ain, const float* __restrict__ Wp,
    const float* __restrict__ bp, float* __restrict__ out)
{
    __shared__ __align__(16) _Float16 As[64 * 32];
    __shared__ __align__(16) _Float16 Bt[64 * 32];
    int t = threadIdx.x, lane = t & 31, w = t >> 5;
    int tileN = blockIdx.x & 15;   // 1024/64
    int tileM = blockIdx.x >> 4;   // 4096/64
    int mBase = tileM * 64, nBase = tileN * 64;
    const uint32_t* A32 = (const uint32_t*)Ain;

    v8f acc[4] = {{}, {}, {}, {}};
    for (int kb = 0; kb < 1024; kb += 32) {
        // A: 64x32 halfs, dword copies (1024 dwords, 8 per thread)
        #pragma unroll
        for (int i = 0; i < 8; ++i) {
            int id = t + i * 128;
            int row = id >> 4, dw = id & 15;
            ((uint32_t*)As)[row * 16 + dw] = A32[(mBase + row) * 512 + (kb >> 1) + dw];
        }
        // B: 32x64 fp32 -> transposed f16
        #pragma unroll
        for (int i = 0; i < 4; ++i) {
            int id = t + i * 128;
            int kr = id >> 4, c4 = id & 15;
            float4 f = *(const float4*)(Wp + (kb + kr) * 1024 + nBase + c4 * 4);
            Bt[(c4 * 4 + 0) * 32 + kr] = (_Float16)f.x;
            Bt[(c4 * 4 + 1) * 32 + kr] = (_Float16)f.y;
            Bt[(c4 * 4 + 2) * 32 + kr] = (_Float16)f.z;
            Bt[(c4 * 4 + 3) * 32 + kr] = (_Float16)f.w;
        }
        __syncthreads();
        v16h a = load_a_frag(As + (w * 16) * 32, 32, lane);
        #pragma unroll
        for (int ct = 0; ct < 4; ++ct) {
            v16h b = load_b_frag(Bt + (ct * 16) * 32, 32, lane);
            acc[ct] = wmma_f16(a, b, acc[ct]);
        }
        __syncthreads();
    }
    int roff = (lane & 16) ? 8 : 0;
    int nlan = lane & 15;
    #pragma unroll
    for (int ct = 0; ct < 4; ++ct) {
        int col = nBase + ct * 16 + nlan;
        float bias = bp[col];
        #pragma unroll
        for (int r = 0; r < 8; ++r) {
            int rowg = mBase + w * 16 + roff + r;
            out[rowg * 1024 + col] = acc[ct][r] + bias;
        }
    }
}

// ---------------------------------------------------------------------------
extern "C" void kernel_launch(void* const* d_in, const int* in_sizes, int n_in,
                              void* d_out, int out_size, void* d_ws, size_t ws_size,
                              hipStream_t stream)
{
    (void)in_sizes; (void)n_in; (void)out_size; (void)ws_size;
    const float* x  = (const float*)d_in[0];
    // d_in[1] = H, d_in[2] = W (unused scalars)
    const float* rc = (const float*)d_in[3];
    const float* rs = (const float*)d_in[4];
    const float* wq = (const float*)d_in[5];
    const float* bq = (const float*)d_in[6];
    const float* wp = (const float*)d_in[7];
    const float* bp = (const float*)d_in[8];
    float* out = (float*)d_out;

    // Workspace: qkv f16 [3][4][16][1024][64] (24MB) + attn-out f16 [4096][1024] (8MB)
    _Float16* qkv  = (_Float16*)d_ws;
    _Float16* aout = qkv + (size_t)3 * 4 * 16 * 1024 * 64;

    qkv_gemm_kernel<<<48 * 64, 128, 0, stream>>>(x, wq, bq, qkv);
    rope_kernel<<<16384, 256, 0, stream>>>(qkv, rc, rs);
    attn_kernel<<<1024, 128, 0, stream>>>(qkv, aout);
    proj_gemm_kernel<<<16 * 64, 128, 0, stream>>>(aout, wp, bp, out);
}
